// ChunkTriangleAttentionStartingNode_858993459585
// MI455X (gfx1250) — matrix operand order repaired and checked by reading, not compiled, not run on hardware
//
#include <hip/hip_runtime.h>

#define NN   384
#define DP   128
#define NH   4
#define CC   32
#define HCC  128
#define MTOT (NN*NN)

typedef __attribute__((ext_vector_type(16))) __bf16       v16bf;
typedef __attribute__((ext_vector_type(8)))  float        v8f;
typedef __attribute__((ext_vector_type(4)))  unsigned int v4u;

union FragB16 { v16bf v; v4u q[2]; };

__device__ __forceinline__ v8f wmma_bf16f32(v16bf a, v16bf b, v8f c) {
  // D = A(16x32 bf16) * B(32x16 bf16) + C(16x16 f32)
  return __builtin_amdgcn_wmma_f32_16x16x32_bf16(false, a, false, b, (short)0, c, false, false);
}

__device__ __forceinline__ float red_max16(float t) {
  t = fmaxf(t, __shfl_xor(t, 1));
  t = fmaxf(t, __shfl_xor(t, 2));
  t = fmaxf(t, __shfl_xor(t, 4));
  t = fmaxf(t, __shfl_xor(t, 8));
  return t;
}
__device__ __forceinline__ float red_sum16(float t) {
  t += __shfl_xor(t, 1);
  t += __shfl_xor(t, 2);
  t += __shfl_xor(t, 4);
  t += __shfl_xor(t, 8);
  return t;
}

// ---------------------------------------------------------------------------
// pass0: weights -> bf16, transposed so WMMA B-fragments are contiguous.
// WT rows: [0,384)   = W_qkv columns (q cols pre-scaled by C^-0.5), inner = d
//          [384,512) = W_gate columns,                              inner = d
//          [512,640) = W_o output cols (d_out),                     inner = e
// ---------------------------------------------------------------------------
__global__ __launch_bounds__(256) void tri_pass0(
    const float* __restrict__ Wqkv, const float* __restrict__ Wg,
    const float* __restrict__ Wo, __bf16* __restrict__ WT)
{
  const int idx = blockIdx.x * 256 + threadIdx.x;
  if (idx >= 640 * DP) return;
  const int row = idx >> 7, d = idx & 127;
  float v;
  if (row < 384) {
    v = Wqkv[(size_t)d * 384 + row];
    if (row < 128) v *= 0.17677669529663687f;   // C^-0.5 folded into q
  } else if (row < 512) {
    v = Wg[(size_t)d * HCC + (row - 384)];
  } else {
    v = Wo[(size_t)d * DP + (row - 512)];       // WT_o[dout][e] = W_o[e][dout]
  }
  WT[idx] = (__bf16)v;
}

// ---------------------------------------------------------------------------
// pass1: LayerNorm + qkv/gate projections (WMMA) + triangle-bias projection.
// One block = 64 pair positions. 256 threads = 8 waves.
// ---------------------------------------------------------------------------
__global__ __launch_bounds__(256) void tri_pass1(
    const float* __restrict__ Zr, const float* __restrict__ lnw,
    const float* __restrict__ lnb, const float* __restrict__ Wb,
    const float* __restrict__ gbias, const __bf16* __restrict__ WT,
    __bf16* __restrict__ qws, __bf16* __restrict__ kws,
    __bf16* __restrict__ vtws, float* __restrict__ gatews,
    float* __restrict__ biasws)
{
  __shared__ __bf16 Zt[64][DP];   // 16 KB normalized tile (bf16)
  const int tid   = threadIdx.x;
  const int mbase = blockIdx.x * 64;

  // ---- LayerNorm: 4 lanes per row, 32 channels each ----
  {
    const int r = tid >> 2, qd = tid & 3;
    const int m = mbase + r;
    const float* zrow = Zr + (size_t)m * DP + qd * 32;
    float x[32];
#pragma unroll
    for (int u = 0; u < 8; ++u) {
      float4 f = ((const float4*)zrow)[u];
      x[4*u+0] = f.x; x[4*u+1] = f.y; x[4*u+2] = f.z; x[4*u+3] = f.w;
    }
    float s = 0.f, ss = 0.f;
#pragma unroll
    for (int u = 0; u < 32; ++u) { s += x[u]; ss += x[u] * x[u]; }
    s  += __shfl_xor(s, 1);  s  += __shfl_xor(s, 2);
    ss += __shfl_xor(ss, 1); ss += __shfl_xor(ss, 2);
    const float mean = s * (1.f / DP);
    const float var  = ss * (1.f / DP) - mean * mean;
    const float rstd = rsqrtf(var + 1e-5f);

    float bh0 = 0.f, bh1 = 0.f, bh2 = 0.f, bh3 = 0.f;
#pragma unroll
    for (int u = 0; u < 32; ++u) {
      const int d = qd * 32 + u;
      const float y = (x[u] - mean) * rstd * lnw[d] + lnb[d];
      Zt[r][d] = (__bf16)y;
      const float4 wb = ((const float4*)Wb)[d];   // W_b[d][0..3]
      bh0 += y * wb.x; bh1 += y * wb.y; bh2 += y * wb.z; bh3 += y * wb.w;
    }
    bh0 += __shfl_xor(bh0, 1); bh0 += __shfl_xor(bh0, 2);
    bh1 += __shfl_xor(bh1, 1); bh1 += __shfl_xor(bh1, 2);
    bh2 += __shfl_xor(bh2, 1); bh2 += __shfl_xor(bh2, 2);
    bh3 += __shfl_xor(bh3, 1); bh3 += __shfl_xor(bh3, 2);
    const float bsel = (qd == 0) ? bh0 : (qd == 1) ? bh1 : (qd == 2) ? bh2 : bh3;
    biasws[(size_t)qd * MTOT + m] = bsel;   // bias[h][q*N + k]
  }
  __syncthreads();

  // ---- Projection GEMM: 64 rows x 512 cols, K = 128 ----
  const int wave = tid >> 5, lane = tid & 31, half = lane >> 4, l16 = lane & 15;
  for (int nt = 0; nt < 4; ++nt) {
    const int ncol = (wave * 4 + nt) * 16;      // 0..511
    v8f acc[4];
    const v8f vzero = {0.f,0.f,0.f,0.f,0.f,0.f,0.f,0.f};
#pragma unroll
    for (int mt = 0; mt < 4; ++mt) acc[mt] = vzero;

#pragma unroll
    for (int kk = 0; kk < 4; ++kk) {
      FragB16 B;   // B[k][n] = WT[ncol+l16][K], K = kk*32 + half*8+{0..7} and +16
      const __bf16* bp = WT + (size_t)(ncol + l16) * DP + kk * 32;
      B.q[0] = *(const v4u*)(bp + half * 8); B.q[1] = *(const v4u*)(bp + 16 + half * 8);
#pragma unroll
      for (int mt = 0; mt < 4; ++mt) {
        FragB16 A;   // A lane: K = kk*32 + half*8+{0..7} and +16
        A.q[0] = *(const v4u*)(&Zt[mt * 16 + l16][kk * 32 + half * 8]);
        A.q[1] = *(const v4u*)(&Zt[mt * 16 + l16][kk * 32 + 16 + half * 8]);
        acc[mt] = wmma_bf16f32(A.v, B.v, acc[mt]);
      }
    }
    // epilogue: route columns to q / k / vT / gate
#pragma unroll
    for (int mt = 0; mt < 4; ++mt) {
#pragma unroll
      for (int rr = 0; rr < 8; ++rr) {
        const int   gm  = mbase + mt * 16 + rr + 8 * half;
        const int   col = ncol + l16;
        const float a   = acc[mt][rr];
        if (col < 128) {
          qws[(size_t)gm * HCC + col] = (__bf16)a;
        } else if (col < 256) {
          kws[(size_t)gm * HCC + (col - 128)] = (__bf16)a;
        } else if (col < 384) {
          const int ca = col - 256, h = ca >> 5, c = ca & 31;
          const int i = gm / NN, j = gm - i * NN;
          vtws[((size_t)(i * NH + h) * CC + c) * NN + j] = (__bf16)a;
        } else {
          const int e = col - 384;
          gatews[(size_t)gm * HCC + e] = 1.f / (1.f + __expf(-(a + gbias[e])));
        }
      }
    }
  }
}

// ---------------------------------------------------------------------------
// pass2: flash attention per (i, h). One wave = one 16-query tile.
// grid = (3 q-blocks, 4 heads, 384 rows i), 8 waves/block.
// ---------------------------------------------------------------------------
__global__ __launch_bounds__(256) void tri_pass2(
    const __bf16* __restrict__ qws, const __bf16* __restrict__ kws,
    const __bf16* __restrict__ vtws, const float* __restrict__ gatews,
    const float* __restrict__ biasws, const float* __restrict__ Zmask,
    __bf16* __restrict__ waws)
{
  __shared__ __bf16 Pt[8][16 * 32];   // 1 KB / wave P staging
  const int i = blockIdx.z, h = blockIdx.y;
  const int tid = threadIdx.x, wave = tid >> 5, lane = tid & 31;
  const int half = lane >> 4, l16 = lane & 15;
  const int qbase = blockIdx.x * 128 + wave * 16;

  FragB16 Aq;   // q[16 x 32] A-fragment, resident across k-loop
  {
    const __bf16* qp = qws + ((size_t)(i * NN + qbase + l16) * HCC + h * CC);
    Aq.q[0] = *(const v4u*)(qp + half * 8);
    Aq.q[1] = *(const v4u*)(qp + 16 + half * 8);
  }

  const v8f vzero = {0.f,0.f,0.f,0.f,0.f,0.f,0.f,0.f};
  v8f O0 = vzero, O1 = vzero;
  float rmax[8], rsum[8];
#pragma unroll
  for (int rr = 0; rr < 8; ++rr) { rmax[rr] = -1e30f; rsum[rr] = 0.f; }

  const float* bq = biasws + (size_t)h * MTOT;
  __bf16* pt = &Pt[wave][0];

  for (int kc = 0; kc < NN / 32; ++kc) {
    const int k0 = kc * 32;
    if (kc + 1 < NN / 32) {   // prefetch next chunk (global_prefetch_b8)
      __builtin_prefetch(kws + ((size_t)(i * NN + k0 + 32 + l16) * HCC + h * CC), 0, 1);
      __builtin_prefetch(vtws + ((size_t)((i * NH + h) * CC + l16) * NN + k0 + 32), 0, 1);
    }
    FragB16 B0, B1;   // B[c][kcol] = k[kcol][c]; lane = kcol, K = c contiguous
    {
      const __bf16* kp = kws + ((size_t)(i * NN + k0 + l16) * HCC + h * CC);
      B0.q[0] = *(const v4u*)(kp + half * 8); B0.q[1] = *(const v4u*)(kp + 16 + half * 8);
    }
    {
      const __bf16* kp = kws + ((size_t)(i * NN + k0 + 16 + l16) * HCC + h * CC);
      B1.q[0] = *(const v4u*)(kp + half * 8); B1.q[1] = *(const v4u*)(kp + 16 + half * 8);
    }
    v8f S0 = wmma_bf16f32(Aq.v, B0.v, vzero);
    v8f S1 = wmma_bf16f32(Aq.v, B1.v, vzero);

    const float mb0 = (Zmask[(size_t)i * NN + k0 + l16] - 1.f) * 1e9f;
    const float mb1 = (Zmask[(size_t)i * NN + k0 + 16 + l16] - 1.f) * 1e9f;

#pragma unroll
    for (int rr = 0; rr < 8; ++rr) {
      const int    qrow = qbase + rr + 8 * half;
      const size_t bi   = (size_t)qrow * NN + k0 + l16;
      S0[rr] += mb0 + bq[bi];
      S1[rr] += mb1 + bq[bi + 16];
    }
    // online softmax update (row reductions across the 16-lane half)
#pragma unroll
    for (int rr = 0; rr < 8; ++rr) {
      const float tm = red_max16(fmaxf(S0[rr], S1[rr]));
      const float nm = fmaxf(rmax[rr], tm);
      const float sc = __expf(rmax[rr] - nm);
      rmax[rr] = nm;
      const float p0 = __expf(S0[rr] - nm);
      const float p1 = __expf(S1[rr] - nm);
      S0[rr] = p0; S1[rr] = p1;
      rsum[rr] = rsum[rr] * sc + red_sum16(p0 + p1);
      O0[rr] *= sc; O1[rr] *= sc;
    }
    // C-frag -> A-frag re-layout through wave-private LDS (DS ops in-order per wave)
#pragma unroll
    for (int rr = 0; rr < 8; ++rr) {
      const int row = rr + 8 * half;
      pt[row * 32 + l16]      = (__bf16)S0[rr];
      pt[row * 32 + 16 + l16] = (__bf16)S1[rr];
    }
    FragB16 Ap;
    Ap.q[0] = *(const v4u*)(&pt[l16 * 32 + half * 8]);
    Ap.q[1] = *(const v4u*)(&pt[l16 * 32 + 16 + half * 8]);
    FragB16 Bv0, Bv1;   // B[k][c] = vT[(i,h,c)][k]; lane = c, K = k contiguous
    {
      const __bf16* vp = vtws + ((size_t)((i * NH + h) * CC + l16) * NN + k0);
      Bv0.q[0] = *(const v4u*)(vp + half * 8); Bv0.q[1] = *(const v4u*)(vp + 16 + half * 8);
    }
    {
      const __bf16* vp = vtws + ((size_t)((i * NH + h) * CC + 16 + l16) * NN + k0);
      Bv1.q[0] = *(const v4u*)(vp + half * 8); Bv1.q[1] = *(const v4u*)(vp + 16 + half * 8);
    }
    O0 = wmma_bf16f32(Ap.v, Bv0.v, O0);
    O1 = wmma_bf16f32(Ap.v, Bv1.v, O1);
  }
  // normalize, gate, store wa (bf16)
#pragma unroll
  for (int rr = 0; rr < 8; ++rr) {
    const int    qrow = qbase + rr + 8 * half;
    const size_t base = (size_t)(i * NN + qrow) * HCC + h * CC;
    const float  inv  = 1.f / rsum[rr];
    waws[base + l16]      = (__bf16)(O0[rr] * inv * gatews[base + l16]);
    waws[base + 16 + l16] = (__bf16)(O1[rr] * inv * gatews[base + 16 + l16]);
  }
}

// ---------------------------------------------------------------------------
// pass3: out = (gate*wa) @ W_o + out_bias + Z_raw. One wave = 16 rows x 128 cols.
// ---------------------------------------------------------------------------
__global__ __launch_bounds__(256) void tri_pass3(
    const __bf16* __restrict__ waws, const __bf16* __restrict__ WTo,
    const float* __restrict__ Zr, const float* __restrict__ obias,
    float* __restrict__ out)
{
  const int tid = threadIdx.x, wave = tid >> 5, lane = tid & 31;
  const int half = lane >> 4, l16 = lane & 15;
  const int mbase = blockIdx.x * 128 + wave * 16;

  const v8f vzero = {0.f,0.f,0.f,0.f,0.f,0.f,0.f,0.f};
  v8f acc[8];
#pragma unroll
  for (int nt = 0; nt < 8; ++nt) acc[nt] = vzero;

#pragma unroll
  for (int kk = 0; kk < 4; ++kk) {
    FragB16 A;
    const __bf16* ap = waws + ((size_t)(mbase + l16) * HCC + kk * 32);
    A.q[0] = *(const v4u*)(ap + half * 8);
    A.q[1] = *(const v4u*)(ap + 16 + half * 8);
#pragma unroll
    for (int nt = 0; nt < 8; ++nt) {
      FragB16 B;
      const __bf16* bp = WTo + ((size_t)(nt * 16 + l16) * DP + kk * 32);
      B.q[0] = *(const v4u*)(bp + half * 8); B.q[1] = *(const v4u*)(bp + 16 + half * 8);
      acc[nt] = wmma_bf16f32(A.v, B.v, acc[nt]);
    }
  }
#pragma unroll
  for (int nt = 0; nt < 8; ++nt) {
#pragma unroll
    for (int rr = 0; rr < 8; ++rr) {
      const int gm = mbase + rr + 8 * half;
      const int d  = nt * 16 + l16;
      out[(size_t)gm * DP + d] = Zr[(size_t)gm * DP + d] + acc[nt][rr] + obias[d];
    }
  }
}

// ---------------------------------------------------------------------------
extern "C" void kernel_launch(void* const* d_in, const int* in_sizes, int n_in,
                              void* d_out, int out_size, void* d_ws, size_t ws_size,
                              hipStream_t stream)
{
  (void)in_sizes; (void)n_in; (void)out_size; (void)ws_size;
  const float* Z_raw    = (const float*)d_in[0];
  const float* Z_mask   = (const float*)d_in[1];
  const float* ln_w     = (const float*)d_in[2];
  const float* ln_b     = (const float*)d_in[3];
  const float* W_b      = (const float*)d_in[4];
  const float* W_qkv    = (const float*)d_in[5];
  const float* W_gate   = (const float*)d_in[6];
  const float* g_bias   = (const float*)d_in[7];
  const float* W_o      = (const float*)d_in[8];
  const float* out_bias = (const float*)d_in[9];
  float* out = (float*)d_out;

  char* ws = (char*)d_ws;
  constexpr size_t SZ_WT   = 640ull * 128 * 2;            // 163840 (256-aligned)
  constexpr size_t SZ_BF   = (size_t)MTOT * HCC * 2;      // 37748736
  constexpr size_t OFF_Q   = SZ_WT;
  constexpr size_t OFF_K   = OFF_Q + SZ_BF;
  constexpr size_t OFF_VT  = OFF_K + SZ_BF;
  constexpr size_t OFF_G   = OFF_VT + SZ_BF;
  constexpr size_t SZ_G    = (size_t)MTOT * HCC * 4;      // 75497472
  constexpr size_t OFF_B   = OFF_G + SZ_G;
  constexpr size_t SZ_B    = (size_t)NH * MTOT * 4;       // 2359296
  constexpr size_t OFF_WA  = OFF_B + SZ_B;                // total ~229 MB

  __bf16* WT    = (__bf16*)(ws);
  __bf16* qws   = (__bf16*)(ws + OFF_Q);
  __bf16* kws   = (__bf16*)(ws + OFF_K);
  __bf16* vtws  = (__bf16*)(ws + OFF_VT);
  float*  gatews= (float*)(ws + OFF_G);
  float*  biasws= (float*)(ws + OFF_B);
  __bf16* waws  = (__bf16*)(ws + OFF_WA);

  tri_pass0<<<dim3(320), dim3(256), 0, stream>>>(W_qkv, W_gate, W_o, WT);
  tri_pass1<<<dim3(MTOT / 64), dim3(256), 0, stream>>>(
      Z_raw, ln_w, ln_b, W_b, g_bias, WT, qws, kws, vtws, gatews, biasws);
  tri_pass2<<<dim3(3, NH, NN), dim3(256), 0, stream>>>(
      qws, kws, vtws, gatews, biasws, Z_mask, waws);
  tri_pass3<<<dim3(MTOT / 128), dim3(256), 0, stream>>>(
      waws, WT + (size_t)512 * DP, Z_raw, out_bias, out);
}
